// ProNodeBlock_87548613362081
// MI455X (gfx1250) — matrix operand order repaired and checked by reading, not compiled, not run on hardware
//
#include <hip/hip_runtime.h>
#include <hip/hip_bf16.h>

// ---------------- types / helpers ----------------
typedef float   v8f    __attribute__((ext_vector_type(8)));
typedef float   f32x8  __attribute__((ext_vector_type(8)));
typedef float   f32x16 __attribute__((ext_vector_type(16)));
typedef __bf16  v16bf  __attribute__((ext_vector_type(16)));
typedef __bf16  bf16x8 __attribute__((ext_vector_type(8)));

#define PITCH 136   // bf16 stage row pitch (elements): 272B rows, 16B aligned, conflict-free

__device__ __forceinline__ v16bf cat8(bf16x8 a, bf16x8 b) {
  return __builtin_shufflevector(a, b, 0, 1, 2, 3, 4, 5, 6, 7, 8, 9, 10, 11, 12, 13, 14, 15);
}
__device__ __forceinline__ v8f wmma_bf16(v16bf a, v16bf b, v8f c) {
  return __builtin_amdgcn_wmma_f32_16x16x32_bf16(false, a, false, b, (short)0, c, false, false);
}

// A fragment (16x32) from f32 row-major source. rowptr = this lane's row (M = lane&15).
// ISA 16-bit A layout: hi half-wave holds K 8..15 / 24..31 of the chunk.
__device__ __forceinline__ v16bf load_A_f32(const float* rowptr, int c, int hi) {
  const int k0 = 32 * c + 8 * hi;
  f32x8 lo = *(const f32x8*)(rowptr + k0);
  f32x8 up = *(const f32x8*)(rowptr + k0 + 16);
  return cat8(__builtin_convertvector(lo, bf16x8), __builtin_convertvector(up, bf16x8));
}

// A fragment from bf16 row-major source (global gather row or LDS stage row).
__device__ __forceinline__ v16bf load_A_bf16(const __bf16* rowptr, int c, int hi) {
  const int k0 = 32 * c + 8 * hi;
  bf16x8 a = *(const bf16x8*)(rowptr + k0);
  bf16x8 b = *(const bf16x8*)(rowptr + k0 + 16);
  return cat8(a, b);
}

// B fragment (32x16): weights torch-style [out][K] row-major bf16; lane's column = weight row,
// so each lane reads two 16-byte runs. hi half-wave holds K 8..15 / 24..31 of the chunk.
__device__ __forceinline__ v16bf load_B(const __bf16* w, int K, int ntile, int c, int lane) {
  const int col = ntile * 16 + (lane & 15);
  const int kb  = 32 * c + 8 * (lane >> 4);
  const bf16x8* p = (const bf16x8*)(w + (size_t)col * K + kb);
  return cat8(p[0], p[2]);
}

// Store f32 C-layout acc (VGPR v -> row v+8*hi, col = ntile*16+(lane&15)) to bf16 LDS stage.
__device__ __forceinline__ void store_C_stage(__bf16* st, int ntile, v8f acc, int lane) {
  const int colo = ntile * 16 + (lane & 15);
  const int hi = lane >> 4;
  bf16x8 h = __builtin_convertvector(acc, bf16x8);
#pragma unroll
  for (int v = 0; v < 8; v++) st[(v + 8 * hi) * PITCH + colo] = h[v];
}

// ---------------- prep kernels ----------------
__global__ void k_f32_to_bf16(const float* __restrict__ s, __bf16* __restrict__ d, int n) {
  int i = blockIdx.x * blockDim.x + threadIdx.x;
  if (i < n) d[i] = (__bf16)s[i];
}
__global__ void k_zero(float* __restrict__ p, int n) {
  int i = blockIdx.x * blockDim.x + threadIdx.x;
  if (i < n) p[i] = 0.f;
}

// ---------------- node kernel: h_node = MLP(x) (bf16 out), cent = x@cent_w.T + b ----------------
__global__ void __launch_bounds__(128) k_node(
    const float* __restrict__ x,
    const __bf16* __restrict__ nw1, const float* __restrict__ nb1,
    const __bf16* __restrict__ nw2, const float* __restrict__ nb2,
    const __bf16* __restrict__ cw,  const float* __restrict__ cb,
    __bf16* __restrict__ h_node_bf, float* __restrict__ cent, int N) {
  __shared__ __bf16 stage[4][16 * PITCH];
  const int lane = threadIdx.x & 31, wid = threadIdx.x >> 5;
  const int tile = blockIdx.x * 4 + wid;
  if (tile * 16 >= N) return;
  const int r = lane & 15, hi = lane >> 4, nbase = tile * 16;
  const float* xrow = x + (size_t)(nbase + r) * 128;

  v16bf ax[4];
#pragma unroll
  for (int c = 0; c < 4; c++) ax[c] = load_A_f32(xrow, c, hi);

  __bf16* st = stage[wid];
  // layer 1: zero-C WMMA chain, bias+relu fused in epilogue -> stage
#pragma unroll
  for (int n = 0; n < 8; n++) {
    v8f acc = {};
#pragma unroll
    for (int c = 0; c < 4; c++) acc = wmma_bf16(ax[c], load_B(nw1, 128, n, c, lane), acc);
    const float bv = nb1[n * 16 + r];
#pragma unroll
    for (int v = 0; v < 8; v++) acc[v] = fmaxf(acc[v] + bv, 0.f);
    store_C_stage(st, n, acc, lane);
  }
  __builtin_amdgcn_wave_barrier();
  v16bf at[4];
#pragma unroll
  for (int c = 0; c < 4; c++) at[c] = load_A_bf16(st + r * PITCH, c, hi);
  // layer 2 -> h_node (bf16)
#pragma unroll
  for (int n = 0; n < 8; n++) {
    v8f acc = {};
#pragma unroll
    for (int c = 0; c < 4; c++) acc = wmma_bf16(at[c], load_B(nw2, 128, n, c, lane), acc);
    const float bv = nb2[n * 16 + r];
#pragma unroll
    for (int v = 0; v < 8; v++) acc[v] += bv;
    bf16x8 h = __builtin_convertvector(acc, bf16x8);
#pragma unroll
    for (int v = 0; v < 8; v++)
      h_node_bf[(size_t)(nbase + v + 8 * hi) * 128 + n * 16 + r] = h[v];
  }
  // centroid linear -> cent (f32)
#pragma unroll
  for (int n = 0; n < 8; n++) {
    v8f acc = {};
#pragma unroll
    for (int c = 0; c < 4; c++) acc = wmma_bf16(ax[c], load_B(cw, 128, n, c, lane), acc);
    const float bv = cb[n * 16 + r];
#pragma unroll
    for (int v = 0; v < 8; v++)
      cent[(size_t)(nbase + v + 8 * hi) * 128 + n * 16 + r] = acc[v] + bv;
  }
}

// ---------------- edge kernel ----------------
__global__ void __launch_bounds__(128) k_edge(
    const float* __restrict__ edge_attr, const int* __restrict__ ei,
    const __bf16* __restrict__ x_bf, const __bf16* __restrict__ h_node_bf,
    const __bf16* __restrict__ ew1, const float* __restrict__ eb1,
    const __bf16* __restrict__ ew2, const float* __restrict__ eb2,
    const __bf16* __restrict__ gw1, const float* __restrict__ gb1,
    const __bf16* __restrict__ gw2, const float* __restrict__ gb2,
    const __bf16* __restrict__ mw,  const float* __restrict__ mb,
    float* __restrict__ aggr, int E) {
  __shared__ __bf16 sm[4][3][16 * PITCH];
  const int lane = threadIdx.x & 31, wid = threadIdx.x >> 5;
  const int tile = blockIdx.x * 4 + wid;
  if (tile * 16 >= E) return;
  const int r = lane & 15, hi = lane >> 4, e0 = tile * 16;

  const int colr = ei[E + e0 + r];                 // col index of this lane's A-row
  int rows[8];
#pragma unroll
  for (int v = 0; v < 8; v++) rows[v] = ei[e0 + v + 8 * hi];  // row index of C rows v+8*hi

  const float* earow = edge_attr + (size_t)(e0 + r) * 64;
  v16bf ae[2];
#pragma unroll
  for (int c = 0; c < 2; c++) ae[c] = load_A_f32(earow, c, hi);
  const __bf16* xrow = x_bf + (size_t)colr * 128;
  v16bf axc[4];
#pragma unroll
  for (int c = 0; c < 4; c++) axc[c] = load_A_bf16(xrow, c, hi);

  __bf16* stA = sm[wid][0];
  __bf16* stB = sm[wid][1];
  __bf16* stC = sm[wid][2];

  // gate layer1 (K=192: edge_attr ++ x[col]) + bias+relu -> stA
#pragma unroll
  for (int n = 0; n < 8; n++) {
    v8f acc = {};
    acc = wmma_bf16(ae[0], load_B(gw1, 192, n, 0, lane), acc);
    acc = wmma_bf16(ae[1], load_B(gw1, 192, n, 1, lane), acc);
#pragma unroll
    for (int c = 0; c < 4; c++) acc = wmma_bf16(axc[c], load_B(gw1, 192, n, 2 + c, lane), acc);
    const float bv = gb1[n * 16 + r];
#pragma unroll
    for (int v = 0; v < 8; v++) acc[v] = fmaxf(acc[v] + bv, 0.f);
    store_C_stage(stA, n, acc, lane);
  }
  __builtin_amdgcn_wave_barrier();
  v16bf t1[4];
#pragma unroll
  for (int c = 0; c < 4; c++) t1[c] = load_A_bf16(stA + r * PITCH, c, hi);
  // gate layer2 + bias + sigmoid -> stB
#pragma unroll
  for (int n = 0; n < 8; n++) {
    v8f acc = {};
#pragma unroll
    for (int c = 0; c < 4; c++) acc = wmma_bf16(t1[c], load_B(gw2, 128, n, c, lane), acc);
    const float bv = gb2[n * 16 + r];
#pragma unroll
    for (int v = 0; v < 8; v++) acc[v] = 1.f / (1.f + __expf(-(acc[v] + bv)));
    store_C_stage(stB, n, acc, lane);
  }
  __builtin_amdgcn_wave_barrier();
  // edge layer1 (K=64) + bias+relu -> stA (reuse)
#pragma unroll
  for (int n = 0; n < 8; n++) {
    v8f acc = {};
    acc = wmma_bf16(ae[0], load_B(ew1, 64, n, 0, lane), acc);
    acc = wmma_bf16(ae[1], load_B(ew1, 64, n, 1, lane), acc);
    const float bv = eb1[n * 16 + r];
#pragma unroll
    for (int v = 0; v < 8; v++) acc[v] = fmaxf(acc[v] + bv, 0.f);
    store_C_stage(stA, n, acc, lane);
  }
  __builtin_amdgcn_wave_barrier();
#pragma unroll
  for (int c = 0; c < 4; c++) t1[c] = load_A_bf16(stA + r * PITCH, c, hi);
  // edge layer2 + bias -> stC (h_edge)
#pragma unroll
  for (int n = 0; n < 8; n++) {
    v8f acc = {};
#pragma unroll
    for (int c = 0; c < 4; c++) acc = wmma_bf16(t1[c], load_B(ew2, 128, n, c, lane), acc);
    const float bv = eb2[n * 16 + r];
#pragma unroll
    for (int v = 0; v < 8; v++) acc[v] += bv;
    store_C_stage(stC, n, acc, lane);
  }
  __builtin_amdgcn_wave_barrier();
  // t = h_edge * h_node[col]   (A layout, bf16; vector cvt + f32 multiply)
  const __bf16* hrow = h_node_bf + (size_t)colr * 128;
  v16bf tt[4];
#pragma unroll
  for (int c = 0; c < 4; c++) {
    f32x16 he = __builtin_convertvector(load_A_bf16(stC + r * PITCH, c, hi), f32x16);
    f32x16 hn = __builtin_convertvector(load_A_bf16(hrow, c, hi), f32x16);
    tt[c] = __builtin_convertvector(he * hn, v16bf);
  }
  // msg GEMM, bias + gate scale, atomic scatter (segment_sum)
#pragma unroll
  for (int n = 0; n < 8; n++) {
    v8f acc = {};
#pragma unroll
    for (int c = 0; c < 4; c++) acc = wmma_bf16(tt[c], load_B(mw, 128, n, c, lane), acc);
    const float bv = mb[n * 16 + r];
#pragma unroll
    for (int v = 0; v < 8; v++) {
      float g = (float)stB[(v + 8 * hi) * PITCH + n * 16 + r];
      atomicAdd(aggr + (size_t)rows[v] * 128 + n * 16 + r, (acc[v] + bv) * g);
    }
  }
}

// ---------------- output kernel: LN(cent+aggr) -> relu -> @out_w.T + out_b ----------------
__global__ void __launch_bounds__(128) k_out(
    const float* __restrict__ cent, const float* __restrict__ aggr,
    const float* __restrict__ lng, const float* __restrict__ lnb,
    const __bf16* __restrict__ ow, const float* __restrict__ ob,
    float* __restrict__ out, int N) {
  const int lane = threadIdx.x & 31, wid = threadIdx.x >> 5;
  const int tile = blockIdx.x * 4 + wid;
  if (tile * 16 >= N) return;
  const int r = lane & 15, hi = lane >> 4, nbase = tile * 16;
  const float* crow = cent + (size_t)(nbase + r) * 128;
  const float* arow = aggr + (size_t)(nbase + r) * 128;

  f32x16 vals[4];
  float s1 = 0.f, s2 = 0.f;
#pragma unroll
  for (int c = 0; c < 4; c++) {
    const int k0 = 32 * c + 8 * hi;
    f32x8 lo = *(const f32x8*)(crow + k0)      + *(const f32x8*)(arow + k0);
    f32x8 up = *(const f32x8*)(crow + k0 + 16) + *(const f32x8*)(arow + k0 + 16);
#pragma unroll
    for (int j = 0; j < 8; j++) {
      vals[c][j] = lo[j]; vals[c][8 + j] = up[j];
      s1 += lo[j] + up[j];
      s2 += lo[j] * lo[j] + up[j] * up[j];
    }
  }
  // lanes L and L^16 jointly hold one row: combine partials
  s1 += __shfl_xor(s1, 16, 32);
  s2 += __shfl_xor(s2, 16, 32);
  const float mean = s1 * (1.f / 128.f);
  const float var  = s2 * (1.f / 128.f) - mean * mean;
  const float rstd = rsqrtf(var + 1e-5f);

  v16bf a[4];
#pragma unroll
  for (int c = 0; c < 4; c++) {
    const int k0 = 32 * c + 8 * hi;
    f32x16 y;
#pragma unroll
    for (int j = 0; j < 16; j++) {
      const int k = k0 + (j < 8 ? j : 8 + j);
      y[j] = fmaxf((vals[c][j] - mean) * rstd * lng[k] + lnb[k], 0.f);
    }
    a[c] = __builtin_convertvector(y, v16bf);
  }
#pragma unroll
  for (int n = 0; n < 8; n++) {
    v8f acc = {};
#pragma unroll
    for (int c = 0; c < 4; c++) acc = wmma_bf16(a[c], load_B(ow, 128, n, c, lane), acc);
    const float bv = ob[n * 16 + r];
#pragma unroll
    for (int v = 0; v < 8; v++)
      out[(size_t)(nbase + v + 8 * hi) * 128 + n * 16 + r] = acc[v] + bv;
  }
}

// ---------------- host ----------------
extern "C" void kernel_launch(void* const* d_in, const int* in_sizes, int n_in,
                              void* d_out, int out_size, void* d_ws, size_t ws_size,
                              hipStream_t stream) {
  const float* x         = (const float*)d_in[0];
  const int*   ei        = (const int*)d_in[2];
  const float* edge_attr = (const float*)d_in[3];
  const float* node_w1 = (const float*)d_in[4];  const float* node_b1 = (const float*)d_in[5];
  const float* node_w2 = (const float*)d_in[6];  const float* node_b2 = (const float*)d_in[7];
  const float* edge_w1 = (const float*)d_in[8];  const float* edge_b1 = (const float*)d_in[9];
  const float* edge_w2 = (const float*)d_in[10]; const float* edge_b2 = (const float*)d_in[11];
  const float* msg_w   = (const float*)d_in[12]; const float* msg_b   = (const float*)d_in[13];
  const float* gate_w1 = (const float*)d_in[14]; const float* gate_b1 = (const float*)d_in[15];
  const float* gate_w2 = (const float*)d_in[16]; const float* gate_b2 = (const float*)d_in[17];
  const float* cent_w  = (const float*)d_in[18]; const float* cent_b  = (const float*)d_in[19];
  const float* ln_g    = (const float*)d_in[20]; const float* ln_b    = (const float*)d_in[21];
  const float* out_w   = (const float*)d_in[22]; const float* out_b   = (const float*)d_in[23];

  const int N = in_sizes[0] / 128;
  const int E = in_sizes[3] / 64;

  char* ws = (char*)d_ws;
  auto take = [&](size_t bytes) { char* p = ws; ws += (bytes + 255) & ~(size_t)255; return p; };
  __bf16* x_bf      = (__bf16*)take((size_t)N * 128 * 2);
  __bf16* h_node_bf = (__bf16*)take((size_t)N * 128 * 2);
  float*  cent      = (float*)take((size_t)N * 128 * 4);
  float*  aggr      = (float*)take((size_t)N * 128 * 4);
  __bf16* w_nw1 = (__bf16*)take(128 * 128 * 2);
  __bf16* w_nw2 = (__bf16*)take(128 * 128 * 2);
  __bf16* w_ew1 = (__bf16*)take(128 * 64 * 2);
  __bf16* w_ew2 = (__bf16*)take(128 * 128 * 2);
  __bf16* w_gw1 = (__bf16*)take(128 * 192 * 2);
  __bf16* w_gw2 = (__bf16*)take(128 * 128 * 2);
  __bf16* w_mw  = (__bf16*)take(128 * 128 * 2);
  __bf16* w_cw  = (__bf16*)take(128 * 128 * 2);
  __bf16* w_ow  = (__bf16*)take(128 * 128 * 2);

  auto cvt = [&](const float* s, __bf16* d, int n) {
    k_f32_to_bf16<<<(n + 255) / 256, 256, 0, stream>>>(s, d, n);
  };
  cvt(x, x_bf, N * 128);
  cvt(node_w1, w_nw1, 128 * 128); cvt(node_w2, w_nw2, 128 * 128);
  cvt(edge_w1, w_ew1, 128 * 64);  cvt(edge_w2, w_ew2, 128 * 128);
  cvt(gate_w1, w_gw1, 128 * 192); cvt(gate_w2, w_gw2, 128 * 128);
  cvt(msg_w,   w_mw,  128 * 128); cvt(cent_w,  w_cw,  128 * 128);
  cvt(out_w,   w_ow,  128 * 128);

  k_zero<<<(N * 128 + 255) / 256, 256, 0, stream>>>(aggr, N * 128);

  const int ntiles = N / 16, etiles = E / 16;
  k_node<<<(ntiles + 3) / 4, 128, 0, stream>>>(x, w_nw1, node_b1, w_nw2, node_b2,
                                               w_cw, cent_b, h_node_bf, cent, N);
  k_edge<<<(etiles + 3) / 4, 128, 0, stream>>>(edge_attr, ei, x_bf, h_node_bf,
                                               w_ew1, edge_b1, w_ew2, edge_b2,
                                               w_gw1, gate_b1, w_gw2, gate_b2,
                                               w_mw, msg_b, aggr, E);
  k_out<<<(ntiles + 3) / 4, 128, 0, stream>>>(cent, aggr, ln_g, ln_b, w_ow, out_b,
                                              (float*)d_out, N);
}
